// VMambaB_79663053406391
// MI455X (gfx1250) — matrix operand order repaired and checked by reading, not compiled, not run on hardware
//
#include <hip/hip_runtime.h>
#include <hip/hip_bf16.h>
#include <math.h>

// ---------------------------------------------------------------------------
// Types for WMMA
// ---------------------------------------------------------------------------
typedef __attribute__((ext_vector_type(16))) _Float16 v16h;
typedef __attribute__((ext_vector_type(8)))  _Float16 v8h;
typedef __attribute__((ext_vector_type(4)))  _Float16 v4h;
typedef __attribute__((ext_vector_type(8)))  float    v8f;

#define TILE_M 64
#define TILE_N 64
#define TILE_K 64   // two 32-deep WMMA K-steps per staged tile

__device__ __forceinline__ float gelu_f(float v) {
    return 0.5f * v * (1.0f + erff(v * 0.70710678118654752f));
}
__device__ __forceinline__ float silu_f(float v) {
    return v / (1.0f + __expf(-v));
}

// ---------------------------------------------------------------------------
// WMMA GEMM: C = act( A(MxK) @ B(KxN) + bias ) + resid
// A, B row-major fp32 -> f16 tiles in LDS -> v_wmma_f32_16x16x32_f16.
// A tile stored row-major [m][k]; B tile stored TRANSPOSED [n][k] so both
// fragment gathers are contiguous 16B runs (ds_load_b128).
// act: 0=none, 1=gelu, 2=silu.
// ---------------------------------------------------------------------------
__global__ __launch_bounds__(128) void gemm_wmma_kernel(
    const float* __restrict__ A, const float* __restrict__ B,
    const float* __restrict__ bias, const float* __restrict__ resid,
    float* __restrict__ C, int M, int N, int K, int act)
{
    // row pitch 72 halves = 144 B (multiple of 16 B -> aligned v8h chunks)
    __shared__ _Float16 Ash[TILE_M][TILE_K + 8];
    __shared__ _Float16 Bsh[TILE_N][TILE_K + 8];

    const int tid  = threadIdx.x;
    const int lane = tid & 31;
    const int wave = tid >> 5;           // 0..3
    const int wm   = (wave >> 1) * 32;   // 0 / 32
    const int wn   = (wave & 1) * 32;    // 0 / 32

    const int block_m = blockIdx.y * TILE_M;
    const int block_n = blockIdx.x * TILE_N;

    const bool m_full = (block_m + TILE_M) <= M;
    const bool n_full = (block_n + TILE_N) <= N;

    v8f acc[2][2] = {};

    for (int k0 = 0; k0 < K; k0 += TILE_K) {
        const bool k_full = (k0 + TILE_K) <= K;

        // ---- stage A tile (64x64) ----
        if (m_full && k_full) {
            // vector fast path: float4 global loads, 8B LDS stores
            for (int i = tid; i < TILE_M * (TILE_K / 4); i += 128) {
                int r = i / (TILE_K / 4), cq = i % (TILE_K / 4);
                float4 v = *(const float4*)&A[(size_t)(block_m + r) * K + k0 + cq * 4];
                v4h h; h[0] = (_Float16)v.x; h[1] = (_Float16)v.y;
                       h[2] = (_Float16)v.z; h[3] = (_Float16)v.w;
                *(v4h*)&Ash[r][cq * 4] = h;
            }
        } else {
            for (int i = tid; i < TILE_M * TILE_K; i += 128) {
                int r = i / TILE_K, c = i % TILE_K;
                int gm = block_m + r, gk = k0 + c;
                int cm = gm < M ? gm : M - 1;
                int ck = gk < K ? gk : K - 1;
                float v = A[(size_t)cm * K + ck];
                v = (gm < M && gk < K) ? v : 0.0f;
                Ash[r][c] = (_Float16)v;
            }
        }
        // ---- stage B tile (64x64), transposed into LDS [n][k] ----
        if (n_full && k_full) {
            for (int i = tid; i < TILE_K * (TILE_N / 4); i += 128) {
                int r = i / (TILE_N / 4), cq = i % (TILE_N / 4);
                float4 v = *(const float4*)&B[(size_t)(k0 + r) * N + block_n + cq * 4];
                Bsh[cq * 4 + 0][r] = (_Float16)v.x;
                Bsh[cq * 4 + 1][r] = (_Float16)v.y;
                Bsh[cq * 4 + 2][r] = (_Float16)v.z;
                Bsh[cq * 4 + 3][r] = (_Float16)v.w;
            }
        } else {
            for (int i = tid; i < TILE_K * TILE_N; i += 128) {
                int r = i / TILE_N, c = i % TILE_N;
                int gk = k0 + r, gn = block_n + c;
                int ck = gk < K ? gk : K - 1;
                int cn = gn < N ? gn : N - 1;
                float v = B[(size_t)ck * N + cn];
                v = (gk < K && gn < N) ? v : 0.0f;
                Bsh[c][r] = (_Float16)v;
            }
        }
        // hint next K tiles toward caches (global_prefetch_b8 path)
        if (k0 + TILE_K < K) {
            __builtin_prefetch(&A[(size_t)block_m * K + k0 + TILE_K], 0, 1);
            __builtin_prefetch(&B[(size_t)(k0 + TILE_K) * N + block_n], 0, 1);
        }
        __syncthreads();

        // ---- compute: 2 K-steps x (2x2) WMMA tiles ----
        // A frag (16x32 f16): lane m = lane&15, elem i -> K = ks*32 + (i/8)*16 + (lane/16)*8 + i%8
        //   -> two contiguous 8-half (16B) runs at ks*32 + hi*8 and ks*32 + 16 + hi*8
        // B frag (32x16 f16): lane n = lane&15, elem i -> K = ks*32 + (i/8)*16 + (lane/16)*8 + i%8
        //   -> two contiguous runs at ks*32 + hi*8 (+16)   [Bsh is [n][k]]
        const int am = lane & 15;
        const int hi = lane >> 4;
#pragma unroll
        for (int ks = 0; ks < 2; ++ks) {
            v16h afrag[2], bfrag[2];
#pragma unroll
            for (int t = 0; t < 2; ++t) {
                const _Float16* ap = &Ash[wm + t * 16 + am][ks * 32 + hi * 8];
                v8h alo = *(const v8h*)ap;
                v8h ahi = *(const v8h*)(ap + 16);
                afrag[t] = __builtin_shufflevector(alo, ahi,
                    0, 1, 2, 3, 4, 5, 6, 7, 8, 9, 10, 11, 12, 13, 14, 15);
                const _Float16* bp = &Bsh[wn + t * 16 + am][ks * 32 + hi * 8];
                v8h blo = *(const v8h*)bp;
                v8h bhi = *(const v8h*)(bp + 16);
                bfrag[t] = __builtin_shufflevector(blo, bhi,
                    0, 1, 2, 3, 4, 5, 6, 7, 8, 9, 10, 11, 12, 13, 14, 15);
            }
#pragma unroll
            for (int tm = 0; tm < 2; ++tm)
#pragma unroll
                for (int tn = 0; tn < 2; ++tn)
                    acc[tm][tn] = __builtin_amdgcn_wmma_f32_16x16x32_f16(
                        false, afrag[tm], false, bfrag[tn],
                        (short)0, acc[tm][tn], false, false);
        }
        __syncthreads();
    }

    // Epilogue. C/D layout: vgpr r -> row = r + 8*(lane>=16), col = lane&15.
    const int hi2 = lane >> 4, nn = lane & 15;
#pragma unroll
    for (int tm = 0; tm < 2; ++tm)
#pragma unroll
        for (int tn = 0; tn < 2; ++tn)
#pragma unroll
            for (int r = 0; r < 8; ++r) {
                int gm = block_m + wm + tm * 16 + r + (hi2 << 3);
                int gn = block_n + wn + tn * 16 + nn;
                if (gm < M && gn < N) {
                    float v = acc[tm][tn][r];
                    if (bias)  v += bias[gn];
                    if (act == 1) v = gelu_f(v);
                    else if (act == 2) v = silu_f(v);
                    if (resid) v += resid[(size_t)gm * N + gn];
                    C[(size_t)gm * N + gn] = v;
                }
            }
}

// ---------------------------------------------------------------------------
// Stem conv1: NCHW fp32 input (3,224,224), k3 s2 p1 -> NHWC (112,112,OC)
// ---------------------------------------------------------------------------
__global__ void conv_stem1_kernel(const float* __restrict__ in,
                                  const float* __restrict__ wgt,
                                  const float* __restrict__ bias,
                                  float* __restrict__ out,
                                  int H, int W, int OH, int OW, int OC)
{
    int idx = blockIdx.x * blockDim.x + threadIdx.x;
    int total = OH * OW * OC;
    if (idx >= total) return;
    int oc = idx % OC, p = idx / OC, ox = p % OW, oy = p / OW;
    float acc = bias[oc];
    for (int ky = 0; ky < 3; ++ky) {
        int iy = oy * 2 + ky - 1;
        if (iy < 0 || iy >= H) continue;
        for (int kx = 0; kx < 3; ++kx) {
            int ix = ox * 2 + kx - 1;
            if (ix < 0 || ix >= W) continue;
            for (int ic = 0; ic < 3; ++ic)
                acc += in[(size_t)ic * H * W + (size_t)iy * W + ix] *
                       wgt[((oc * 3 + ic) * 3 + ky) * 3 + kx];
        }
    }
    out[(size_t)p * OC + oc] = acc;
}

// ---------------------------------------------------------------------------
// Generic conv k3 p1: NHWC in -> NHWC out, weights OIHW
// ---------------------------------------------------------------------------
__global__ void conv_nhwc_kernel(const float* __restrict__ in,
                                 const float* __restrict__ wgt,
                                 const float* __restrict__ bias,
                                 float* __restrict__ out,
                                 int H, int W, int IC, int OH, int OW, int OC,
                                 int stride)
{
    int idx = blockIdx.x * blockDim.x + threadIdx.x;
    int total = OH * OW * OC;
    if (idx >= total) return;
    int oc = idx % OC, p = idx / OC, ox = p % OW, oy = p / OW;
    float acc = bias[oc];
    for (int ky = 0; ky < 3; ++ky) {
        int iy = oy * stride + ky - 1;
        if (iy < 0 || iy >= H) continue;
        for (int kx = 0; kx < 3; ++kx) {
            int ix = ox * stride + kx - 1;
            if (ix < 0 || ix >= W) continue;
            const float* ip = in + ((size_t)iy * W + ix) * IC;
            const float* wp = wgt + ((size_t)oc * IC * 9) + ky * 3 + kx;
            for (int ic = 0; ic < IC; ++ic)
                acc += ip[ic] * wp[(size_t)ic * 9];
        }
    }
    out[(size_t)p * OC + oc] = acc;
}

// ---------------------------------------------------------------------------
// LayerNorm over last dim; rows = grid.x, block 256
// ---------------------------------------------------------------------------
__global__ __launch_bounds__(256) void layernorm_kernel(
    const float* __restrict__ in, float* __restrict__ out,
    const float* __restrict__ g, const float* __restrict__ b, int C)
{
    int row = blockIdx.x;
    const float* xr = in + (size_t)row * C;
    float* yr = out + (size_t)row * C;
    float s = 0.f, s2 = 0.f;
    for (int c = threadIdx.x; c < C; c += 256) { float v = xr[c]; s += v; s2 += v * v; }
    __shared__ float rs[256], rq[256];
    rs[threadIdx.x] = s; rq[threadIdx.x] = s2; __syncthreads();
    for (int off = 128; off > 0; off >>= 1) {
        if (threadIdx.x < off) { rs[threadIdx.x] += rs[threadIdx.x + off];
                                 rq[threadIdx.x] += rq[threadIdx.x + off]; }
        __syncthreads();
    }
    float mu = rs[0] / C;
    float var = rq[0] / C - mu * mu;
    float rstd = rsqrtf(var + 1e-5f);
    for (int c = threadIdx.x; c < C; c += 256)
        yr[c] = (xr[c] - mu) * rstd * g[c] + b[c];
}

__global__ void gelu_kernel(float* __restrict__ x, int n)
{
    int i = blockIdx.x * blockDim.x + threadIdx.x;
    if (i < n) x[i] = gelu_f(x[i]);
}

// ---------------------------------------------------------------------------
// Depthwise conv 3x3 s1 p1 + SiLU. Input xi = rows (L, ld) at channel cols
// [0, di). Output channel-first (di, L).
// ---------------------------------------------------------------------------
__global__ void dwconv_silu_kernel(const float* __restrict__ xi, int ld,
                                   const float* __restrict__ cw,
                                   const float* __restrict__ cb,
                                   float* __restrict__ xc,
                                   int h, int w, int di)
{
    int L = h * w;
    int idx = blockIdx.x * blockDim.x + threadIdx.x;
    if (idx >= di * L) return;
    int d = idx / L, i = idx % L, rr = i / w, cc = i % w;
    float acc = cb[d];
    for (int ky = 0; ky < 3; ++ky) {
        int iy = rr + ky - 1;
        if (iy < 0 || iy >= h) continue;
        for (int kx = 0; kx < 3; ++kx) {
            int ix = cc + kx - 1;
            if (ix < 0 || ix >= w) continue;
            acc += xi[((size_t)iy * w + ix) * ld + d] * cw[d * 9 + ky * 3 + kx];
        }
    }
    xc[(size_t)d * L + i] = silu_f(acc);
}

// ---------------------------------------------------------------------------
// Cross-scan: xc (di, L) -> xs (4, di, L)
// ---------------------------------------------------------------------------
__global__ void cross_scan_kernel(const float* __restrict__ xc,
                                  float* __restrict__ xs,
                                  int di, int h, int w)
{
    int L = h * w;
    int idx = blockIdx.x * blockDim.x + threadIdx.x;
    if (idx >= di * L) return;
    int d = idx / L, i = idx % L, rr = i / w, cc = i % w;
    float v = xc[(size_t)d * L + i];
    int j = cc * h + rr;
    size_t dl = (size_t)d * L;
    size_t kL = (size_t)di * L;
    xs[0 * kL + dl + i]           = v;
    xs[1 * kL + dl + j]           = v;
    xs[2 * kL + dl + (L - 1 - i)] = v;
    xs[3 * kL + dl + (L - 1 - j)] = v;
}

// ---------------------------------------------------------------------------
// dts = softplus(dts + dt_b[k][d]) in place; dts (4, di, L)
// ---------------------------------------------------------------------------
__global__ void softplus_bias_kernel(float* __restrict__ dts,
                                     const float* __restrict__ dtb,
                                     int di, int L)
{
    int idx = blockIdx.x * blockDim.x + threadIdx.x;
    int total = 4 * di * L;
    if (idx >= total) return;
    int k = idx / (di * L);
    int d = (idx / L) % di;
    float v = dts[idx] + dtb[k * di + d];
    dts[idx] = (v > 20.0f) ? v : log1pf(__expf(v));
}

// ---------------------------------------------------------------------------
// Selective scan: one thread per (k, channel); 16-state SSM in registers.
// A = -exp(A_log) computed on the fly.
// ---------------------------------------------------------------------------
__global__ void selective_scan_kernel(const float* __restrict__ xs,
                                      const float* __restrict__ dts,
                                      const float* __restrict__ xdbl,
                                      const float* __restrict__ Alog,
                                      const float* __restrict__ Ds,
                                      float* __restrict__ ys,
                                      int di, int L, int r0, int Cx)
{
    int ch = blockIdx.x * blockDim.x + threadIdx.x;
    if (ch >= 4 * di) return;
    int k = ch / di, d = ch % di;
    const float* u  = xs  + ((size_t)k * di + d) * L;
    const float* dt = dts + ((size_t)k * di + d) * L;
    const float* Bp = xdbl + ((size_t)k * Cx + r0) * L;
    const float* Cp = xdbl + ((size_t)k * Cx + r0 + 16) * L;
    const float* Ak = Alog + ((size_t)k * di + d) * 16;
    float*       yp = ys  + ((size_t)k * di + d) * L;
    float a[16], hst[16];
#pragma unroll
    for (int n = 0; n < 16; ++n) { a[n] = -__expf(Ak[n]); hst[n] = 0.0f; }
    float D = Ds[k * di + d];
    for (int t = 0; t < L; ++t) {
        float dtv = dt[t], uv = u[t], du = dtv * uv;
        float y = 0.0f;
#pragma unroll
        for (int n = 0; n < 16; ++n) {
            hst[n] = __expf(dtv * a[n]) * hst[n] + du * Bp[(size_t)n * L + t];
            y += hst[n] * Cp[(size_t)n * L + t];
        }
        yp[t] = y + D * uv;
    }
}

// ---------------------------------------------------------------------------
// Cross-merge + LayerNorm(di) + silu(z) gate. ys (4,di,L) -> out (L,di)
// z read from xz rows at columns [di, 2*di).
// ---------------------------------------------------------------------------
__global__ __launch_bounds__(256) void merge_ln_silu_kernel(
    const float* __restrict__ ys, const float* __restrict__ xz,
    const float* __restrict__ g, const float* __restrict__ b,
    float* __restrict__ out, int di, int h, int w)
{
    int L = h * w;
    int i = blockIdx.x;
    int rr = i / w, cc = i % w;
    int j = cc * h + rr;
    size_t kL = (size_t)di * L;
    float* orow = out + (size_t)i * di;
    float s = 0.f, s2 = 0.f;
    for (int d = threadIdx.x; d < di; d += 256) {
        size_t dl = (size_t)d * L;
        float v = ys[0 * kL + dl + i] + ys[2 * kL + dl + (L - 1 - i)]
                + ys[1 * kL + dl + j] + ys[3 * kL + dl + (L - 1 - j)];
        orow[d] = v;
        s += v; s2 += v * v;
    }
    __shared__ float rs[256], rq[256];
    rs[threadIdx.x] = s; rq[threadIdx.x] = s2; __syncthreads();
    for (int off = 128; off > 0; off >>= 1) {
        if (threadIdx.x < off) { rs[threadIdx.x] += rs[threadIdx.x + off];
                                 rq[threadIdx.x] += rq[threadIdx.x + off]; }
        __syncthreads();
    }
    float mu = rs[0] / di;
    float var = rq[0] / di - mu * mu;
    float rstd = rsqrtf(var + 1e-5f);
    const float* zrow = xz + (size_t)i * (2 * di) + di;
    for (int d = threadIdx.x; d < di; d += 256) {
        float v = (orow[d] - mu) * rstd * g[d] + b[d];
        orow[d] = v * silu_f(zrow[d]);
    }
}

// ---------------------------------------------------------------------------
// Mean pool over rows: x (L, C) -> feat (C)
// ---------------------------------------------------------------------------
__global__ void meanpool_kernel(const float* __restrict__ x,
                                float* __restrict__ feat, int L, int C)
{
    int c = blockIdx.x * blockDim.x + threadIdx.x;
    if (c >= C) return;
    float s = 0.f;
    for (int i = 0; i < L; ++i) s += x[(size_t)i * C + c];
    feat[c] = s / L;
}

__global__ void head_gemv_kernel(const float* __restrict__ feat,
                                 const float* __restrict__ wh,
                                 const float* __restrict__ bh,
                                 float* __restrict__ logits, int K, int N)
{
    int j = blockIdx.x * blockDim.x + threadIdx.x;
    if (j >= N) return;
    float s = bh[j];
    for (int i = 0; i < K; ++i) s += feat[i] * wh[(size_t)i * N + j];
    logits[j] = s;
}

__global__ __launch_bounds__(256) void softmax_kernel(
    const float* __restrict__ logits, float* __restrict__ out, int N)
{
    __shared__ float red[256];
    float m = -1e30f;
    for (int i = threadIdx.x; i < N; i += 256) m = fmaxf(m, logits[i]);
    red[threadIdx.x] = m; __syncthreads();
    for (int off = 128; off > 0; off >>= 1) {
        if (threadIdx.x < off) red[threadIdx.x] = fmaxf(red[threadIdx.x], red[threadIdx.x + off]);
        __syncthreads();
    }
    m = red[0]; __syncthreads();
    float s = 0.f;
    for (int i = threadIdx.x; i < N; i += 256) s += __expf(logits[i] - m);
    red[threadIdx.x] = s; __syncthreads();
    for (int off = 128; off > 0; off >>= 1) {
        if (threadIdx.x < off) red[threadIdx.x] += red[threadIdx.x + off];
        __syncthreads();
    }
    float inv = 1.0f / red[0];
    for (int i = threadIdx.x; i < N; i += 256)
        out[i] = __expf(logits[i] - m) * inv;
}

// ---------------------------------------------------------------------------
// Host orchestration
// ---------------------------------------------------------------------------
static inline void launch_gemm(const float* A, const float* B, const float* bias,
                               const float* resid, float* C, int M, int N, int K,
                               int act, hipStream_t s)
{
    dim3 grid((N + TILE_N - 1) / TILE_N, (M + TILE_M - 1) / TILE_M);
    gemm_wmma_kernel<<<grid, 128, 0, s>>>(A, B, bias, resid, C, M, N, K, act);
}

static inline int ceil_div(int a, int b) { return (a + b - 1) / b; }

extern "C" void kernel_launch(void* const* d_in, const int* in_sizes, int n_in,
                              void* d_out, int out_size, void* d_ws, size_t ws_size,
                              hipStream_t stream)
{
    (void)in_sizes; (void)n_in; (void)out_size; (void)ws_size;
    float* W = (float*)d_ws;

    // bump allocator (units: floats) — peak sized by stage 0 (L=3136, d=128)
    size_t o = 0;
    auto alloc = [&](size_t n) { float* p = W + o; o += n; return p; };
    float* XA   = alloc(401408);   // activation ping (L*d max)
    float* XB   = alloc(401408);   // activation pong / stem conv2 out
    float* T0   = alloc(401408);   // LN scratch / downsample conv out
    float* XZ   = alloc(1605632);  // in_proj out (L, 4d)
    float* XC   = alloc(802816);   // dwconv out (di, L) / stem LN scratch
    float* XS   = alloc(3211264);  // cross-scan (4, di, L) / stem conv1 out
    float* XDBL = alloc(501760);   // x_proj out (4, Cx, L)
    float* DTS  = alloc(3211264);  // dt (4, di, L)
    float* YS   = alloc(3211264);  // scan out (4, di, L)
    float* YMOD = alloc(802816);   // merged+gated (L, di)
    float* HMLP = alloc(1605632);  // MLP hidden (L, 4d)
    float* FEAT = alloc(1024);
    float* LOGI = alloc(1000);

    int cur = 0;
    auto nx = [&]() { return (const float*)d_in[cur++]; };

    const float* x_img = nx();                 // (1,3,224,224) NCHW

    // ---- stem ----
    const float* c1w = nx(); const float* c1b = nx();
    const float* sl1g = nx(); const float* sl1b = nx();
    const float* c2w = nx(); const float* c2b = nx();
    const float* sl2g = nx(); const float* sl2b = nx();

    {
        int OH = 112, OW = 112, OC = 64;
        int tot = OH * OW * OC;
        conv_stem1_kernel<<<ceil_div(tot, 256), 256, 0, stream>>>(
            x_img, c1w, c1b, XS, 224, 224, OH, OW, OC);
        layernorm_kernel<<<OH * OW, 256, 0, stream>>>(XS, XC, sl1g, sl1b, OC);
        gelu_kernel<<<ceil_div(tot, 256), 256, 0, stream>>>(XC, tot);
        int OH2 = 56, OW2 = 56, OC2 = 128;
        int tot2 = OH2 * OW2 * OC2;
        conv_nhwc_kernel<<<ceil_div(tot2, 256), 256, 0, stream>>>(
            XC, c2w, c2b, XB, OH, OW, OC, OH2, OW2, OC2, 2);
        layernorm_kernel<<<OH2 * OW2, 256, 0, stream>>>(XB, XA, sl2g, sl2b, OC2);
    }

    const int DIMS[4]   = {128, 256, 512, 1024};
    const int DEPTHS[4] = {2, 2, 15, 2};
    const int HWS[4]    = {56, 28, 14, 7};

    float* xcur = XA;

    for (int s = 0; s < 4; ++s) {
        int d  = DIMS[s];
        int di = 2 * d;
        int r  = (d + 15) / 16;
        int Cx = r + 32;
        int h = HWS[s], w = HWS[s], L = h * w;

        for (int blk = 0; blk < DEPTHS[s]; ++blk) {
            const float* ln1g = nx(); const float* ln1b = nx();
            const float* ipw  = nx(); const float* ipb  = nx();
            const float* cw   = nx(); const float* cb   = nx();
            const float* xpw  = nx();
            const float* dtw  = nx();
            const float* dtb  = nx();
            const float* alog = nx();
            const float* dsp  = nx();
            const float* olg  = nx(); const float* olb  = nx();
            const float* opw  = nx(); const float* opb  = nx();
            const float* ln2g = nx(); const float* ln2b = nx();
            const float* f1w  = nx(); const float* f1b  = nx();
            const float* f2w  = nx(); const float* f2b  = nx();

            // ss2d branch
            layernorm_kernel<<<L, 256, 0, stream>>>(xcur, T0, ln1g, ln1b, d);
            launch_gemm(T0, ipw, ipb, nullptr, XZ, L, 4 * d, d, 0, stream);
            dwconv_silu_kernel<<<ceil_div(di * L, 256), 256, 0, stream>>>(
                XZ, 2 * di, cw, cb, XC, h, w, di);
            cross_scan_kernel<<<ceil_div(di * L, 256), 256, 0, stream>>>(
                XC, XS, di, h, w);
            for (int k = 0; k < 4; ++k) {
                launch_gemm(xpw + (size_t)k * Cx * di, XS + (size_t)k * di * L,
                            nullptr, nullptr, XDBL + (size_t)k * Cx * L,
                            Cx, L, di, 0, stream);
            }
            for (int k = 0; k < 4; ++k) {
                launch_gemm(dtw + (size_t)k * di * r, XDBL + (size_t)k * Cx * L,
                            nullptr, nullptr, DTS + (size_t)k * di * L,
                            di, L, r, 0, stream);
            }
            softplus_bias_kernel<<<ceil_div(4 * di * L, 256), 256, 0, stream>>>(
                DTS, dtb, di, L);
            selective_scan_kernel<<<ceil_div(4 * di, 128), 128, 0, stream>>>(
                XS, DTS, XDBL, alog, dsp, YS, di, L, r, Cx);
            merge_ln_silu_kernel<<<L, 256, 0, stream>>>(
                YS, XZ, olg, olb, YMOD, di, h, w);
            launch_gemm(YMOD, opw, opb, xcur, xcur, L, d, di, 0, stream);

            // MLP branch
            layernorm_kernel<<<L, 256, 0, stream>>>(xcur, T0, ln2g, ln2b, d);
            launch_gemm(T0, f1w, f1b, nullptr, HMLP, L, 4 * d, d, 1, stream);
            launch_gemm(HMLP, f2w, f2b, xcur, xcur, L, d, 4 * d, 0, stream);
        }

        if (s < 3) {
            const float* dw  = nx(); const float* db = nx();
            const float* dlg = nx(); const float* dlb = nx();
            int d2 = DIMS[s + 1];
            int oh = HWS[s + 1], ow = HWS[s + 1];
            int tot = oh * ow * d2;
            conv_nhwc_kernel<<<ceil_div(tot, 256), 256, 0, stream>>>(
                xcur, dw, db, T0, h, w, d, oh, ow, d2, 2);
            float* xnext = (xcur == XA) ? XB : XA;
            layernorm_kernel<<<oh * ow, 256, 0, stream>>>(T0, xnext, dlg, dlb, d2);
            xcur = xnext;
        }
    }

    // ---- head ----
    const float* hw = nx(); const float* hb = nx();
    meanpool_kernel<<<ceil_div(1024, 256), 256, 0, stream>>>(xcur, FEAT, 49, 1024);
    head_gemv_kernel<<<ceil_div(1000, 256), 256, 0, stream>>>(FEAT, hw, hb, LOGI, 1024, 1000);
    softmax_kernel<<<1, 256, 0, stream>>>(LOGI, (float*)d_out, 1000);
}
